// VARQuantizer_49314814493013
// MI455X (gfx1250) — matrix operand order repaired and checked by reading, not compiled, not run on hardware
//
#include <hip/hip_runtime.h>
#include <hip/hip_bf16.h>

typedef __attribute__((ext_vector_type(16))) __bf16       v16bf;
typedef __attribute__((ext_vector_type(8)))  float        v8f;
typedef __attribute__((ext_vector_type(4)))  unsigned int v4u;
typedef __attribute__((ext_vector_type(4)))  unsigned int u32x4;
typedef __attribute__((ext_vector_type(8)))  int          i32x8;
typedef __attribute__((ext_vector_type(4)))  int          i32x4;

union V16BF { v16bf v; v4u q[2]; unsigned short u[16]; };

#define BB   128
#define CC   32
#define VV   4096
#define HW   256          // 16*16
#define NTOT (BB*CC*HW)   // 1048576
#define CHUNK 256         // codes staged per LDS chunk (16 KB bf16)

__device__ __forceinline__ unsigned short f2bf(float x) {
    unsigned int u = __float_as_uint(x);
    unsigned int r = u + 0x7FFFu + ((u >> 16) & 1u);
    return (unsigned short)(r >> 16);
}

// Issue a TDM 1-D tile load: 16 KB codebook chunk k -> LDS buffer (k&1).
// D# per CDNA5 ISA ch.8: group0 = {flags, lds_addr, global_addr, type=2},
// group1 = {data_size=8B, tensor_dim0, tensor_dim1=1, tile_dim0=2048, stride}.
__device__ __forceinline__ void tdm_load_chunk(const unsigned short* embb, int k,
                                               unsigned lds_base) {
    unsigned long long ga = (unsigned long long)(size_t)embb +
                            (unsigned long long)k * (CHUNK * CC * 2);
    unsigned lds_addr = lds_base + (unsigned)(k & 1) * (CHUNK * CC * 2);
    const unsigned ELEMS = (CHUNK * CC * 2) / 8;     // 2048 x 8-byte elements
    u32x4 g0; i32x8 g1;
    i32x4 g2 = {0, 0, 0, 0}, g3 = {0, 0, 0, 0};
    g0[0] = 1u;                                              // count=1 (valid, user)
    g0[1] = lds_addr;                                        // lds_addr (bytes)
    g0[2] = (unsigned)(ga & 0xFFFFFFFFull);                  // global_addr[31:0]
    g0[3] = (unsigned)((ga >> 32) & 0x01FFFFFFull) | (2u << 30); // addr[56:32] | type=2
    g1[0] = (int)(3u << 16);                                 // data_size = 8B
    g1[1] = (int)((ELEMS & 0xFFFFu) << 16);                  // tensor_dim0[15:0]
    g1[2] = (int)((ELEMS >> 16) | (1u << 16));               // tensor_dim0[31:16] | dim1.lo=1
    g1[3] = (int)((ELEMS & 0xFFFFu) << 16);                  // dim1.hi=0 | tile_dim0=2048
    g1[4] = 0;                                               // tile_dim1/2 unused
    g1[5] = (int)ELEMS;                                      // tensor_dim0_stride
    g1[6] = 0; g1[7] = 0;
#if defined(__clang_major__) && (__clang_major__ >= 23)
    i32x8 g4 = {0, 0, 0, 0, 0, 0, 0, 0};
    __builtin_amdgcn_tensor_load_to_lds(g0, g1, g2, g3, g4, 0);
#else
    __builtin_amdgcn_tensor_load_to_lds(g0, g1, g2, g3, 0);
#endif
}

// ---------------- prep kernels ----------------
__global__ __launch_bounds__(256) void prep_emb(const float* __restrict__ emb,
                                                unsigned short* __restrict__ embb,
                                                float* __restrict__ esq) {
    int v = blockIdx.x * blockDim.x + threadIdx.x;
    if (v >= VV) return;
    float s = 0.f;
    #pragma unroll
    for (int c = 0; c < CC; ++c) {
        float x = emb[v * CC + c];
        s += x * x;
        embb[v * CC + c] = f2bf(x);
    }
    esq[v] = s;
}

// phi_W [4][32o][32i][3][3] -> wbt [4][9(tap)][32o][32i] bf16
__global__ __launch_bounds__(256) void prep_w(const float* __restrict__ w,
                                              unsigned short* __restrict__ wbt) {
    int gid = blockIdx.x * blockDim.x + threadIdx.x;
    if (gid >= 4 * 9 * 32 * 32) return;
    int ci  = gid & 31;
    int co  = (gid >> 5) & 31;
    int tap = (gid >> 10) % 9;
    int k   = gid / 9216;
    wbt[gid] = f2bf(w[(((k * 32 + co) * 32 + ci) * 9) + tap]);
}

__global__ __launch_bounds__(256) void prep_state(const float* __restrict__ f,
                                                  float* __restrict__ f_rest,
                                                  float* __restrict__ f_hat,
                                                  float* __restrict__ loss) {
    int gid = blockIdx.x * blockDim.x + threadIdx.x;
    if (gid < NTOT) { f_rest[gid] = f[gid]; f_hat[gid] = 0.f; }
    if (gid == 0) *loss = 0.f;
}

// ---------------- per-scale: area downsample -> bf16 NHWC rows ----------------
__global__ __launch_bounds__(256) void downsample_area(const float* __restrict__ f_rest,
                                                       int pn,
                                                       unsigned short* __restrict__ rest_bf) {
    int gid = blockIdx.x * blockDim.x + threadIdx.x;
    int n = BB * pn * pn * CC;
    if (gid >= n) return;
    int c   = gid & 31;
    int rem = gid >> 5;
    int px  = rem % pn; rem /= pn;
    int py  = rem % pn;
    int b   = rem / pn;
    int sy = (py * 16) / pn, ey = ((py + 1) * 16 + pn - 1) / pn;
    int sx = (px * 16) / pn, ex = ((px + 1) * 16 + pn - 1) / pn;
    const float* base = f_rest + ((size_t)b * CC + c) * HW;
    float s = 0.f;
    for (int y = sy; y < ey; ++y)
        for (int x = sx; x < ex; ++x) s += base[y * 16 + x];
    s *= 1.0f / (float)((ey - sy) * (ex - sx));
    rest_bf[((size_t)(b * pn + py) * pn + px) * CC + c] = f2bf(s);
}

// ---------------- per-scale: WMMA nearest-neighbor search ----------------
// 256 thr = 8 waves; wave owns 16 rows. Codebook streamed by the TDM into
// double-buffered LDS (16 KB chunks), overlapped with the WMMA scan.
__global__ __launch_bounds__(256) void nn_search(const unsigned short* __restrict__ rest_bf,
                                                 const unsigned short* __restrict__ embb,
                                                 const float* __restrict__ esq,
                                                 int* __restrict__ idx_ws,
                                                 float* __restrict__ idx_out) {
    __shared__ __align__(16) unsigned short lds_emb[2][CHUNK * CC]; // 2 x 16 KB
    __shared__ float lds_esq[2][CHUNK];                              // 2 x 1 KB
    int tid  = threadIdx.x;
    int lane = tid & 31, wave = tid >> 5;
    int hi   = lane >> 4, ln = lane & 15;
    int rowbase = blockIdx.x * 128 + wave * 16;

    // A tile: lane holds row (rowbase+ln), K chunks [hi*8, +8) and [16+hi*8, +8)
    V16BF a;
    const unsigned short* rp = rest_bf + (size_t)(rowbase + ln) * CC;
    a.q[0] = *(const v4u*)(rp + hi * 8);
    a.q[1] = *(const v4u*)(rp + 16 + hi * 8);

    float best_d[8];
    int   best_i[8];
    #pragma unroll
    for (int r = 0; r < 8; ++r) { best_d[r] = __int_as_float(0x7F800000); best_i[r] = 0; }

    const unsigned lds_base = (unsigned)(size_t)(&lds_emb[0][0]);
    const int NCHUNK = VV / CHUNK;                 // 16

    // prologue: stage e_sq chunk 0, launch TDM for chunk 0
    lds_esq[0][tid] = esq[tid];
    if (wave == 0) tdm_load_chunk(embb, 0, lds_base);

    for (int k = 0; k < NCHUNK; ++k) {
        __syncthreads();   // prior compute done -> safe to overwrite buffer (k+1)&1
        if (k + 1 < NCHUNK) {
            lds_esq[(k + 1) & 1][tid] = esq[(k + 1) * CHUNK + tid];
            if (wave == 0) tdm_load_chunk(embb, k + 1, lds_base);
        }
        if (wave == 0) {
            if (k + 1 < NCHUNK) __builtin_amdgcn_s_wait_tensorcnt(1); // chunk k landed
            else                __builtin_amdgcn_s_wait_tensorcnt(0);
        }
        __syncthreads();   // buffer k&1 ready for all waves

        const unsigned short* eb = &lds_emb[k & 1][0];
        const float*          eq = &lds_esq[k & 1][0];
        int cbase = k * CHUNK;
        for (int t = 0; t < CHUNK; t += 32) {   // 2 code tiles / iter hides wmma hazard
            V16BF b0, b1;
            const unsigned short* bp0 = eb + (t + ln) * CC + hi * 8;
            const unsigned short* bp1 = eb + (t + 16 + ln) * CC + hi * 8;
            b0.q[0] = *(const v4u*)(bp0); b0.q[1] = *(const v4u*)(bp0 + 16);
            b1.q[0] = *(const v4u*)(bp1); b1.q[1] = *(const v4u*)(bp1 + 16);
            v8f acc0 = {}, acc1 = {};
            acc0 = __builtin_amdgcn_wmma_f32_16x16x32_bf16(
                       false, a.v, false, b0.v, (short)0, acc0, false, false);
            acc1 = __builtin_amdgcn_wmma_f32_16x16x32_bf16(
                       false, a.v, false, b1.v, (short)0, acc1, false, false);
            float es0 = eq[t + ln], es1 = eq[t + 16 + ln];
            int c0i = cbase + t + ln, c1i = cbase + t + 16 + ln;
            #pragma unroll
            for (int r = 0; r < 8; ++r) {        // row = rowbase + r + hi*8
                float d0 = fmaf(-2.0f, acc0[r], es0);
                if (d0 < best_d[r]) { best_d[r] = d0; best_i[r] = c0i; }
                float d1 = fmaf(-2.0f, acc1[r], es1);
                if (d1 < best_d[r]) { best_d[r] = d1; best_i[r] = c1i; }
            }
        }
    }

    // reduce across the 16 column lanes sharing each row (first-index tie-break)
    #pragma unroll
    for (int r = 0; r < 8; ++r) {
        float d = best_d[r];
        int   i = best_i[r];
        #pragma unroll
        for (int m = 1; m < 16; m <<= 1) {
            float od = __shfl_xor(d, m, 32);
            int   oi = __shfl_xor(i, m, 32);
            bool take = (od < d) || ((od == d) && (oi < i));
            d = take ? od : d;
            i = take ? oi : i;
        }
        if (ln == 0) {
            int row = rowbase + r + hi * 8;
            idx_ws[row]  = i;
            idx_out[row] = (float)i;
        }
    }
}

// ---------------- per-scale: gather codes + bicubic upsample to 16x16 ----------------
__device__ __forceinline__ void cub_w(int i, int pn, float* w, int* ix) {
    const float A = -0.75f;
    float src = (i + 0.5f) * (float)pn * (1.0f / 16.0f) - 0.5f;
    float fl  = floorf(src);
    #pragma unroll
    for (int t = -1; t < 3; ++t) {
        float x = fabsf(src - (fl + (float)t));
        float wt;
        if (x <= 1.f)      wt = (A + 2.f) * x * x * x - (A + 3.f) * x * x + 1.f;
        else if (x < 2.f)  wt = A * x * x * x - 5.f * A * x * x + 8.f * A * x - 4.f * A;
        else               wt = 0.f;
        int id = (int)fl + t;
        id = id < 0 ? 0 : (id > pn - 1 ? pn - 1 : id);
        w[t + 1]  = wt;
        ix[t + 1] = id;
    }
}

__global__ __launch_bounds__(256) void gather_upsample(const int* __restrict__ idx,
                                                       const float* __restrict__ emb,
                                                       int pn,
                                                       float* __restrict__ h_up,          // [B][C][256]
                                                       unsigned short* __restrict__ h_bf) // [B*256][C]
{
    int gid = blockIdx.x * blockDim.x + threadIdx.x;
    if (gid >= NTOT) return;
    int ox = gid & 15, oy = (gid >> 4) & 15, c = (gid >> 8) & 31, b = gid >> 13;
    float wy[4], wx[4]; int iy[4], ix[4];
    cub_w(oy, pn, wy, iy);
    cub_w(ox, pn, wx, ix);
    const int* ib = idx + b * pn * pn;
    float acc = 0.f;
    #pragma unroll
    for (int ty = 0; ty < 4; ++ty) {
        float accx = 0.f;
        #pragma unroll
        for (int tx = 0; tx < 4; ++tx)
            accx += wx[tx] * emb[(size_t)ib[iy[ty] * pn + ix[tx]] * CC + c];
        acc += wy[ty] * accx;
    }
    h_up[gid] = acc;
    h_bf[((size_t)(b * HW + oy * 16 + ox)) * CC + c] = f2bf(acc);
}

// ---------------- per-scale: WMMA 3x3 conv (implicit GEMM) + residual update + loss ---
__global__ __launch_bounds__(256) void phi_conv_update(const unsigned short* __restrict__ h_bf,
                                                       const float* __restrict__ h_up,
                                                       const unsigned short* __restrict__ wb, // [9][32co][32ci]
                                                       const float* __restrict__ bias,
                                                       const float* __restrict__ f,
                                                       float* __restrict__ f_hat,
                                                       float* __restrict__ f_rest,
                                                       float* __restrict__ loss_acc,
                                                       float loss_scale) {
    int tid = threadIdx.x, lane = tid & 31, wave = tid >> 5;
    int hi = lane >> 4, ln = lane & 15;
    __builtin_prefetch(wb, 0, 0);
    int pix0 = blockIdx.x * 128 + wave * 16;   // 16 pixels, within one image
    int b  = pix0 >> 8;
    int p0 = pix0 & 255;
    int py = (p0 + ln) >> 4, px = (p0 + ln) & 15;

    v8f c0 = {}, c1 = {};
    #pragma unroll
    for (int ky = 0; ky < 3; ++ky) {
        #pragma unroll
        for (int kx = 0; kx < 3; ++kx) {
            int sy = py + ky - 1, sx = px + kx - 1;
            V16BF a;
            if (sy >= 0 && sy < 16 && sx >= 0 && sx < 16) {
                const unsigned short* rp = h_bf + ((size_t)(b * HW + sy * 16 + sx)) * CC;
                a.q[0] = *(const v4u*)(rp + hi * 8);
                a.q[1] = *(const v4u*)(rp + 16 + hi * 8);
            } else {
                v4u z = {0u, 0u, 0u, 0u};
                a.q[0] = z; a.q[1] = z;
            }
            int tap = ky * 3 + kx;
            V16BF b0, b1;
            const unsigned short* w0 = wb + ((size_t)tap * 32 + ln) * 32 + hi * 8;
            const unsigned short* w1 = wb + ((size_t)tap * 32 + 16 + ln) * 32 + hi * 8;
            b0.q[0] = *(const v4u*)(w0); b0.q[1] = *(const v4u*)(w0 + 16);
            b1.q[0] = *(const v4u*)(w1); b1.q[1] = *(const v4u*)(w1 + 16);
            c0 = __builtin_amdgcn_wmma_f32_16x16x32_bf16(false, a.v, false, b0.v, (short)0, c0, false, false);
            c1 = __builtin_amdgcn_wmma_f32_16x16x32_bf16(false, a.v, false, b1.v, (short)0, c1, false, false);
        }
    }

    float part = 0.f;
    #pragma unroll
    for (int r = 0; r < 8; ++r) {
        int pp = p0 + r + hi * 8;               // pixel in image
        #pragma unroll
        for (int half = 0; half < 2; ++half) {
            int co = half * 16 + ln;
            float conv = (half ? c1[r] : c0[r]) + bias[co];
            size_t off = ((size_t)b * CC + co) * HW + pp;
            float hval = h_up[off];
            float outv = 0.5f * hval + 0.5f * conv;   // RESI = 0.5
            float fh = f_hat[off] + outv;
            f_hat[off]  = fh;
            f_rest[off] -= outv;
            float diff = fh - f[off];
            part += diff * diff;
        }
    }
    __shared__ float red[256];
    red[tid] = part;
    __syncthreads();
    for (int s = 128; s > 0; s >>= 1) {
        if (tid < s) red[tid] += red[tid + s];
        __syncthreads();
    }
    if (tid == 0) atomicAdd(loss_acc, red[0] * loss_scale);
}

__global__ void finalize(const float* __restrict__ loss, float* __restrict__ out) {
    *out = *loss;
}

// ---------------- host ----------------
extern "C" void kernel_launch(void* const* d_in, const int* in_sizes, int n_in,
                              void* d_out, int out_size, void* d_ws, size_t ws_size,
                              hipStream_t stream) {
    const float* f    = (const float*)d_in[0];
    const float* emb  = (const float*)d_in[1];
    const float* phiW = (const float*)d_in[2];
    const float* phiB = (const float*)d_in[3];
    float* out = (float*)d_out;

    char* ws = (char*)d_ws;
    float*          f_rest  = (float*)(ws + 0);                    // 4 MB
    unsigned short* rest_bf = (unsigned short*)(ws + (4u << 20));  // 2 MB
    unsigned short* embb    = (unsigned short*)(ws + (6u << 20));  // 256 KB
    float*          esq     = (float*)(ws + (6u << 20) + (256u << 10)); // 16 KB
    unsigned short* wbt     = (unsigned short*)(ws + (6u << 20) + (272u << 10)); // 72 KB
    int*            idx_ws  = (int*)(ws + (6u << 20) + (384u << 10));            // 128 KB
    float*          h_up    = (float*)(ws + (7u << 20));           // 4 MB
    unsigned short* h_bf    = (unsigned short*)(ws + (11u << 20)); // 2 MB
    float*          loss    = (float*)(ws + (13u << 20));          // 4 B

    float* f_hat   = out;                 // [0, 1048576)
    float* loss_o  = out + NTOT;          // [1048576]
    float* idx_o   = out + NTOT + 1;      // [1048577, +87040)

    static const int PN[10] = {1, 2, 3, 4, 5, 6, 8, 10, 13, 16};
    static const int KT[10] = {0, 0, 0, 1, 1, 2, 2, 2, 3, 3};
    const float LSCALE = (1.0f + 0.25f) / ((float)NTOT * 10.0f);

    prep_emb  <<<VV / 256, 256, 0, stream>>>(emb, embb, esq);
    prep_w    <<<(4 * 9 * 32 * 32) / 256, 256, 0, stream>>>(phiW, wbt);
    prep_state<<<NTOT / 256, 256, 0, stream>>>(f, f_rest, f_hat, loss);

    int off = 0;
    for (int si = 0; si < 10; ++si) {
        int pn = PN[si];
        int N  = BB * pn * pn;
        int k  = KT[si];
        int nthr = N * CC;
        downsample_area<<<(nthr + 255) / 256, 256, 0, stream>>>(f_rest, pn, rest_bf);
        nn_search      <<<pn * pn, 256, 0, stream>>>(rest_bf, embb, esq, idx_ws, idx_o + off);
        gather_upsample<<<NTOT / 256, 256, 0, stream>>>(idx_ws, emb, pn, h_up, h_bf);
        phi_conv_update<<<(BB * HW) / 128, 256, 0, stream>>>(
            h_bf, h_up, wbt + (size_t)k * 9 * 32 * 32, phiB + k * 32,
            f, f_hat, f_rest, loss, LSCALE);
        off += N;
    }
    finalize<<<1, 1, 0, stream>>>(loss, loss_o);
}
